// GaussianAttention_257698038469
// MI455X (gfx1250) — matrix operand order repaired and checked by reading, not compiled, not run on hardware
//
#include <hip/hip_runtime.h>

// ---------------------------------------------------------------------------
// Gaussian attention (compile-only target: gfx1250 / MI455X, wave32, WMMA,
// async global->LDS staging, double-buffered, branch-free steady state)
// ---------------------------------------------------------------------------

typedef _Float16 hvec8  __attribute__((ext_vector_type(8)));
typedef _Float16 hvec16 __attribute__((ext_vector_type(16)));
typedef float    fvec8  __attribute__((ext_vector_type(8)));

#define BS     16
#define D      256
#define NH     9
#define Q      1024      // 32*32 positions
#define FANIN  2304      // NH*D
#define LNEPS  1e-5f
#define LDA    72        // LDS row stride in f16 (64 payload + 8 pad)

// ---------------------------------------------------------------------------
// Async global -> LDS copy helpers (CDNA5: GLOBAL_LOAD_ASYNC_TO_LDS_B128,
// tracked by ASYNCcnt). LDS dest address = low 32 bits of generic pointer
// (workgroup-relative LDS byte offset).
// ---------------------------------------------------------------------------
__device__ __forceinline__ void async_cp16(void* lds_ptr, const void* gptr) {
  asm volatile("global_load_async_to_lds_b128 %0, %1, off"
               :: "v"((unsigned int)(size_t)lds_ptr),
                  "v"((unsigned long long)(size_t)gptr)
               : "memory");
}
__device__ __forceinline__ void wait_async0() {
#if __has_builtin(__builtin_amdgcn_s_wait_asynccnt)
  __builtin_amdgcn_s_wait_asynccnt(0);
#else
  asm volatile("s_wait_asynccnt 0x0" ::: "memory");
#endif
}

// Stage NROWS x 64 f16 tile (row chunks of 8x16B) into LDS, async, branch-free:
// fixed trip count so no per-lane bound check / exec masking is generated.
template <int NROWS>
__device__ __forceinline__ void stage_tile_async(const _Float16* __restrict__ gbase,
                                                 size_t gstride_f16,
                                                 _Float16* lbase, int tid) {
  constexpr int REPS = NROWS * 8 / 256;       // 16B chunks per thread
#pragma unroll
  for (int rep = 0; rep < REPS; ++rep) {
    const int c = tid + rep * 256;
    const int row = c >> 3, off8 = (c & 7) * 8;
    async_cp16(lbase + row * LDA + off8, gbase + (size_t)row * gstride_f16 + off8);
  }
}

// ---------------------------------------------------------------------------
// WMMA operand builders (CDNA5 16-bit 16x16x32 layouts, wave32)
//   A (16x32): lanes 0-15 hold M=lane, K 0-7 & 16-23; lanes 16-31: K 8-15 & 24-31
//   B (32x16): lanes 0-15 hold N=lane, K 0-7 & 16-23; lanes 16-31: K 8-15 & 24-31
// LDS tiles stored row-major with K contiguous, stride LDA f16.
// ---------------------------------------------------------------------------
__device__ __forceinline__ hvec16 lds_opA(const _Float16* base, int row16, int half) {
  hvec8 lo = *(const hvec8*)(base + row16 * LDA + half * 8);
  hvec8 hi = *(const hvec8*)(base + row16 * LDA + half * 8 + 16);
  return __builtin_shufflevector(lo, hi, 0,1,2,3,4,5,6,7,8,9,10,11,12,13,14,15);
}
__device__ __forceinline__ hvec16 lds_opB(const _Float16* base, int col16, int half) {
  hvec8 lo = *(const hvec8*)(base + col16 * LDA + half * 8);
  hvec8 hi = *(const hvec8*)(base + col16 * LDA + half * 8 + 16);
  return __builtin_shufflevector(lo, hi, 0,1,2,3,4,5,6,7,8,9,10,11,12,13,14,15);
}

// One K=64 step: 2 sub-chunks of K=32, 2x2 WMMA tiles per wave.
__device__ __forceinline__ void mma_step(const _Float16* As, const _Float16* Bs,
                                         int wm, int wn, int half, int l15,
                                         fvec8 acc[2][2]) {
#pragma unroll
  for (int ks = 0; ks < 64; ks += 32) {
    hvec16 Aop[2], Bop[2];
    Aop[0] = lds_opA(As + ks, wm * 32 + l15, half);
    Aop[1] = lds_opA(As + ks, wm * 32 + 16 + l15, half);
    Bop[0] = lds_opB(Bs + ks, wn * 32 + l15, half);
    Bop[1] = lds_opB(Bs + ks, wn * 32 + 16 + l15, half);
#pragma unroll
    for (int mi = 0; mi < 2; ++mi)
#pragma unroll
      for (int ni = 0; ni < 2; ++ni)
        acc[mi][ni] = __builtin_amdgcn_wmma_f32_16x16x32_f16(
            false, Aop[mi], false, Bop[ni], (short)0, acc[mi][ni], false, false);
  }
}

// ---------------------------------------------------------------------------
// 1) LayerNorm: one wave per row of 256 floats
// ---------------------------------------------------------------------------
__global__ __launch_bounds__(256) void k_ln(const float* __restrict__ hid,
                                            const float* __restrict__ gamma,
                                            const float* __restrict__ beta,
                                            float* __restrict__ xln) {
  const int wave = threadIdx.x >> 5, lane = threadIdx.x & 31;
  const int row  = blockIdx.x * 8 + wave;               // 16384 rows total
  const float4* src = (const float4*)(hid + (size_t)row * D);
  const float4* g4  = (const float4*)gamma;
  const float4* b4  = (const float4*)beta;
  float4 v0 = src[lane], v1 = src[lane + 32];
  float s  = v0.x + v0.y + v0.z + v0.w + v1.x + v1.y + v1.z + v1.w;
  float ss = v0.x*v0.x + v0.y*v0.y + v0.z*v0.z + v0.w*v0.w
           + v1.x*v1.x + v1.y*v1.y + v1.z*v1.z + v1.w*v1.w;
#pragma unroll
  for (int off = 16; off; off >>= 1) {
    s  += __shfl_xor(s,  off, 32);
    ss += __shfl_xor(ss, off, 32);
  }
  const float mu = s * (1.0f / D);
  const float var = ss * (1.0f / D) - mu * mu;
  const float rs = rsqrtf(var + LNEPS);
  float4 g0 = g4[lane], g1 = g4[lane + 32];
  float4 be0 = b4[lane], be1 = b4[lane + 32];
  float4 o0, o1;
  o0.x = (v0.x - mu) * rs * g0.x + be0.x;  o0.y = (v0.y - mu) * rs * g0.y + be0.y;
  o0.z = (v0.z - mu) * rs * g0.z + be0.z;  o0.w = (v0.w - mu) * rs * g0.w + be0.w;
  o1.x = (v1.x - mu) * rs * g1.x + be1.x;  o1.y = (v1.y - mu) * rs * g1.y + be1.y;
  o1.z = (v1.z - mu) * rs * g1.z + be1.z;  o1.w = (v1.w - mu) * rs * g1.w + be1.w;
  float4* dst = (float4*)(xln + (size_t)row * D);
  dst[lane] = o0;  dst[lane + 32] = o1;
}

// ---------------------------------------------------------------------------
// 2) Transpose + convert: xln[b][k][d] f32 -> x16t[b][d][k] f16 (64x64 tiles)
// ---------------------------------------------------------------------------
__global__ __launch_bounds__(256) void k_tr(const float* __restrict__ xln,
                                            _Float16* __restrict__ x16t) {
  __shared__ _Float16 tile[64][72];
  const int d0 = blockIdx.x * 64, k0 = blockIdx.y * 64, b = blockIdx.z;
  const int t = threadIdx.x;
  {
    const int r = t >> 2, c0 = (t & 3) * 16;          // k-row local, d-col base
    const float4* src = (const float4*)(xln + ((size_t)(b * Q + k0 + r)) * D + d0 + c0);
#pragma unroll
    for (int u = 0; u < 4; ++u) {
      float4 v = src[u];
      tile[r][c0 + u * 4 + 0] = (_Float16)v.x;
      tile[r][c0 + u * 4 + 1] = (_Float16)v.y;
      tile[r][c0 + u * 4 + 2] = (_Float16)v.z;
      tile[r][c0 + u * 4 + 3] = (_Float16)v.w;
    }
  }
  __syncthreads();
  {
    const int dr = t >> 2, kc0 = (t & 3) * 16;
    struct alignas(32) H16 { _Float16 v[16]; } tmp;
#pragma unroll
    for (int u = 0; u < 16; ++u) tmp.v[u] = tile[kc0 + u][dr];
    uint4* dst = (uint4*)(x16t + ((size_t)(b * D + d0 + dr)) * Q + k0 + kc0);
    dst[0] = ((const uint4*)&tmp)[0];
    dst[1] = ((const uint4*)&tmp)[1];
  }
}

// ---------------------------------------------------------------------------
// 3) value_w f32 [n][k] -> f16 [n][k]
// ---------------------------------------------------------------------------
__global__ __launch_bounds__(256) void k_wcvt(const float* __restrict__ w,
                                              _Float16* __restrict__ w16) {
  const size_t idx = (size_t)blockIdx.x * 256 + threadIdx.x;  // 589824 total
  w16[idx] = (_Float16)w[idx];
}

// ---------------------------------------------------------------------------
// 4) Scores + softmax: one block per (head, query). 1024 keys, 4 per thread.
// ---------------------------------------------------------------------------
__global__ __launch_bounds__(256) void k_softmax(const float* __restrict__ centers,
                                                 const float* __restrict__ spreads,
                                                 float* __restrict__ probs_out,
                                                 _Float16* __restrict__ p16) {
  __shared__ float red[8];
  const int hd = blockIdx.x, q = blockIdx.y;
  const int qi = q >> 5, qj = q & 31;
  const int wave = threadIdx.x >> 5, lane = threadIdx.x & 31;

  const float* S = spreads + hd * 4;
  const float s00 = S[0], s01 = S[1], s10 = S[2], s11 = S[3];
  const float a = s00 * s00 + s01 * s01;
  const float bb = s00 * s10 + s01 * s11;
  const float c = s10 * s10 + s11 * s11;
  const float mu1 = centers[hd * 2 + 0], mu2 = centers[hd * 2 + 1];
  // u = -0.5*[-2(a*mu1+b*mu2), -2(c+mu2+b*mu1), a, c, 2b]  (faithful to ref)
  const float u0 = a * mu1 + bb * mu2;
  const float u1 = c + mu2 + bb * mu1;
  const float u2 = -0.5f * a, u3 = -0.5f * c, u4 = -bb;

  float sc[4];
#pragma unroll
  for (int s = 0; s < 4; ++s) {
    const int key = threadIdx.x + 256 * s;
    const float dk = (float)((key >> 5) - qi);
    const float dl = (float)((key & 31) - qj);
    sc[s] = u0 * dk + u1 * dl + u2 * dk * dk + u3 * dl * dl + u4 * dk * dl;
  }
  float m = fmaxf(fmaxf(sc[0], sc[1]), fmaxf(sc[2], sc[3]));
#pragma unroll
  for (int off = 16; off; off >>= 1) m = fmaxf(m, __shfl_xor(m, off, 32));
  if (lane == 0) red[wave] = m;
  __syncthreads();
  float bm = red[0];
#pragma unroll
  for (int w = 1; w < 8; ++w) bm = fmaxf(bm, red[w]);
  __syncthreads();
  float e[4], es = 0.f;
#pragma unroll
  for (int s = 0; s < 4; ++s) { e[s] = __expf(sc[s] - bm); es += e[s]; }
#pragma unroll
  for (int off = 16; off; off >>= 1) es += __shfl_xor(es, off, 32);
  if (lane == 0) red[wave] = es;
  __syncthreads();
  float tot = 0.f;
#pragma unroll
  for (int w = 0; w < 8; ++w) tot += red[w];
  const float inv = 1.0f / tot;
#pragma unroll
  for (int s = 0; s < 4; ++s) {
    const int key = threadIdx.x + 256 * s;
    const float p = e[s] * inv;
    probs_out[(size_t)(q * NH + hd) * Q + key] = p;            // d_out layout
    p16[((size_t)hd * Q + q) * Q + key] = (_Float16)p;         // [h][q][k]
  }
}

// ---------------------------------------------------------------------------
// 5) Context GEMM: for (b,h): C[q][d] = sum_k P_h[q][k] * X_b[k][d]
//    Block tile 128(M) x 64(N), 8 waves (4x2), wave tile 32x32 = 2x2 WMMA.
//    K=1024 in chunks of 64, double-buffered async LDS staging; pipeline is
//    unrolled two chunks per iteration so buffer identities are static.
// ---------------------------------------------------------------------------
__global__ __launch_bounds__(256) void k_ctx_gemm(const _Float16* __restrict__ p16,
                                                  const _Float16* __restrict__ x16t,
                                                  _Float16* __restrict__ ctx) {
  __shared__ _Float16 As[2][128 * LDA];
  __shared__ _Float16 Bs[2][64 * LDA];
  const int q0 = blockIdx.x * 128, d0 = blockIdx.y * 64;
  const int b = blockIdx.z / NH, hd = blockIdx.z % NH;
  const int tid = threadIdx.x;
  const int wave = tid >> 5, lane = tid & 31;
  const int wm = wave >> 1, wn = wave & 1;
  const int half = lane >> 4, l15 = lane & 15;

  const _Float16* gA = p16 + ((size_t)(hd * Q + q0)) * Q;   // row stride Q
  const _Float16* gB = x16t + ((size_t)(b * D + d0)) * Q;   // row stride Q

  fvec8 acc[2][2] = {};

  stage_tile_async<128>(gA, Q, As[0], tid);
  stage_tile_async<64>(gB, Q, Bs[0], tid);

  for (int kk = 0; kk < Q; kk += 128) {     // 8 iterations, 2 chunks each
    wait_async0();
    __syncthreads();                        // buffer 0 ready
    stage_tile_async<128>(gA + kk + 64, Q, As[1], tid);   // always in range
    stage_tile_async<64>(gB + kk + 64, Q, Bs[1], tid);
    mma_step(As[0], Bs[0], wm, wn, half, l15, acc);

    wait_async0();
    __syncthreads();                        // buffer 1 ready
    if (kk + 128 < Q) {                     // uniform scalar branch (tail only)
      stage_tile_async<128>(gA + kk + 128, Q, As[0], tid);
      stage_tile_async<64>(gB + kk + 128, Q, Bs[0], tid);
    }
    mma_step(As[1], Bs[1], wm, wn, half, l15, acc);
  }

  // D layout: vgpr r, lanes0-15 -> (M=r, N=lane); lanes16-31 -> (M=r+8, N=lane-16)
#pragma unroll
  for (int mi = 0; mi < 2; ++mi)
#pragma unroll
    for (int ni = 0; ni < 2; ++ni) {
      const int col = d0 + wn * 32 + ni * 16 + l15;
#pragma unroll
      for (int r = 0; r < 8; ++r) {
        const int row = q0 + wm * 32 + mi * 16 + half * 8 + r;
        ctx[((size_t)(b * Q + row)) * FANIN + hd * D + col] = (_Float16)acc[mi][ni][r];
      }
    }
}

// ---------------------------------------------------------------------------
// 6) Output GEMM: out[m][n] = sum_k ctx[m][k]*W[n][k] + bias[n] + xln[m][n]
//    M=16384, K=2304, N=256. Same tiling and pipeline (2304/64 = 36 chunks).
// ---------------------------------------------------------------------------
__global__ __launch_bounds__(256) void k_out_gemm(const _Float16* __restrict__ ctx,
                                                  const _Float16* __restrict__ w16,
                                                  const float* __restrict__ bias,
                                                  const float* __restrict__ xln,
                                                  float* __restrict__ out) {
  __shared__ _Float16 As[2][128 * LDA];
  __shared__ _Float16 Bs[2][64 * LDA];
  const int n0 = blockIdx.x * 64, m0 = blockIdx.y * 128;
  const int tid = threadIdx.x;
  const int wave = tid >> 5, lane = tid & 31;
  const int wm = wave >> 1, wn = wave & 1;
  const int half = lane >> 4, l15 = lane & 15;

  const _Float16* gA = ctx + (size_t)m0 * FANIN;            // row stride FANIN
  const _Float16* gB = w16 + (size_t)n0 * FANIN;            // row stride FANIN

  fvec8 acc[2][2] = {};

  stage_tile_async<128>(gA, FANIN, As[0], tid);
  stage_tile_async<64>(gB, FANIN, Bs[0], tid);

  for (int kk = 0; kk < FANIN; kk += 128) { // 18 iterations, 2 chunks each
    wait_async0();
    __syncthreads();
    stage_tile_async<128>(gA + kk + 64, FANIN, As[1], tid);
    stage_tile_async<64>(gB + kk + 64, FANIN, Bs[1], tid);
    mma_step(As[0], Bs[0], wm, wn, half, l15, acc);

    wait_async0();
    __syncthreads();
    if (kk + 128 < FANIN) {
      stage_tile_async<128>(gA + kk + 128, FANIN, As[0], tid);
      stage_tile_async<64>(gB + kk + 128, FANIN, Bs[0], tid);
    }
    mma_step(As[1], Bs[1], wm, wn, half, l15, acc);
  }

#pragma unroll
  for (int mi = 0; mi < 2; ++mi)
#pragma unroll
    for (int ni = 0; ni < 2; ++ni) {
      const int col = n0 + wn * 32 + ni * 16 + l15;
      const float bcol = bias[col];
#pragma unroll
      for (int r = 0; r < 8; ++r) {
        const int row = m0 + wm * 32 + mi * 16 + half * 8 + r;
        out[(size_t)row * D + col] = acc[mi][ni][r] + bcol + xln[(size_t)row * D + col];
      }
    }
}

// ---------------------------------------------------------------------------
// Launcher. Workspace layout (bytes):
//   xln  f32 [16][1024][256]       16,777,216
//   x16t f16 [16][256][1024]        8,388,608
//   p16  f16 [9][1024][1024]       18,874,368
//   w16  f16 [256][2304]            1,179,648
//   ctx  f16 [16][1024][2304]      75,497,472   (total ~120.7 MB)
// ---------------------------------------------------------------------------
extern "C" void kernel_launch(void* const* d_in, const int* in_sizes, int n_in,
                              void* d_out, int out_size, void* d_ws, size_t ws_size,
                              hipStream_t stream) {
  const float* hidden  = (const float*)d_in[0];
  const float* centers = (const float*)d_in[1];
  const float* spreads = (const float*)d_in[2];
  const float* value_w = (const float*)d_in[3];
  const float* value_b = (const float*)d_in[4];
  const float* gamma   = (const float*)d_in[5];
  const float* beta    = (const float*)d_in[6];

  float* out       = (float*)d_out;
  float* probs_out = out + (size_t)BS * Q * D;   // second tuple element

  char* ws = (char*)d_ws;
  constexpr size_t XLN_OFF  = 0;
  constexpr size_t X16T_OFF = XLN_OFF  + (size_t)BS * Q * D * 4;
  constexpr size_t P16_OFF  = X16T_OFF + (size_t)BS * D * Q * 2;
  constexpr size_t W16_OFF  = P16_OFF  + (size_t)NH * Q * Q * 2;
  constexpr size_t CTX_OFF  = W16_OFF  + (size_t)D * FANIN * 2;

  float*    xln  = (float*)(ws + XLN_OFF);
  _Float16* x16t = (_Float16*)(ws + X16T_OFF);
  _Float16* p16  = (_Float16*)(ws + P16_OFF);
  _Float16* w16  = (_Float16*)(ws + W16_OFF);
  _Float16* ctx  = (_Float16*)(ws + CTX_OFF);

  k_ln<<<dim3(2048), dim3(256), 0, stream>>>(hidden, gamma, beta, xln);
  k_tr<<<dim3(4, 16, 16), dim3(256), 0, stream>>>(xln, x16t);
  k_wcvt<<<dim3(FANIN), dim3(256), 0, stream>>>(value_w, w16);
  k_softmax<<<dim3(NH, Q), dim3(256), 0, stream>>>(centers, spreads, probs_out, p16);
  k_ctx_gemm<<<dim3(8, 4, BS * NH), dim3(256), 0, stream>>>(p16, x16t, ctx);
  k_out_gemm<<<dim3(4, 128), dim3(256), 0, stream>>>(ctx, w16, value_b, xln, out);
}
